// CrossAttentionBlock_52862457479217
// MI455X (gfx1250) — compile-verified, matrix operand order checked by reading, not run on hardware
//
#include <hip/hip_runtime.h>

typedef __attribute__((ext_vector_type(16))) _Float16 v16h;
typedef __attribute__((ext_vector_type(8)))  float    v8f;

#define B_   16
#define C_   512
#define CD_  768
#define HW_  4096
#define L_   77
#define LP   80     // padded key rows (5 x 16 tiles)
#define LP2  96     // padded key width for PV (3 x 32 K-chunks)
#define NH   8
#define HD   64
#define GC   16     // channels per group (C/32)

__device__ __forceinline__ v8f wmma16(v16h a, v16h b, v8f c) {
    return __builtin_amdgcn_wmma_f32_16x16x32_f16(false, a, false, b, (short)0, c, false, false);
}

// Fragment loader for A ([M][K] row-major) and B-as-Bt ([N][K] row-major), global or LDS.
// Per ISA 7.12.2: lane<16 -> row lane, K 0..7 & 16..23 ; lane>=16 -> row lane-16, K 8..15 & 24..31.
__device__ __forceinline__ v16h frag16(const _Float16* p, int ld) {
    int lane = threadIdx.x & 31;
    const _Float16* b = p + (size_t)(lane & 15) * ld + ((lane >> 4) << 3);
    v16h f;
#pragma unroll
    for (int i = 0; i < 8; ++i) { f[i] = b[i]; f[i + 8] = b[i + 16]; }
    return f;
}

// f32 source (cond) with clamped row index (L=77 tail tile).
__device__ __forceinline__ v16h frag16_f32c(const float* p, int ld, int rows_left) {
    int lane = threadIdx.x & 31;
    int r = lane & 15;
    if (r >= rows_left) r = rows_left - 1;
    const float* b = p + (size_t)r * ld + ((lane >> 4) << 3);
    v16h f;
#pragma unroll
    for (int i = 0; i < 8; ++i) { f[i] = (_Float16)b[i]; f[i + 8] = (_Float16)b[i + 16]; }
    return f;
}

// Async global->LDS staging (CDNA5 ASYNCcnt path). 16B chunks; cpr = chunks per row.
__device__ __forceinline__ void stage_async(_Float16* dst, const _Float16* src,
                                            int src_ld, int dst_ld, int nchunks, int cpr) {
    unsigned base = (unsigned)(size_t)(void*)dst;   // low 32 bits of generic addr == LDS offset
    for (int i = threadIdx.x; i < nchunks; i += 256) {
        int row = i / cpr;
        int cc  = (i - row * cpr) * 8;
        unsigned long long g = (unsigned long long)(src + (size_t)row * src_ld + cc);
        unsigned l = base + (unsigned)((row * dst_ld + cc) * 2);
        asm volatile("global_load_async_to_lds_b128 %0, %1, off" :: "v"(l), "v"(g) : "memory");
    }
}

__device__ __forceinline__ void wait_async_barrier() {
    asm volatile("s_wait_asynccnt 0" ::: "memory");
    __syncthreads();
}

__global__ void cab_cvt_f16(const float* __restrict__ in, _Float16* __restrict__ out, int n) {
    int i = blockIdx.x * blockDim.x + threadIdx.x;
    if (i < n) out[i] = (_Float16)in[i];
}

__global__ void cab_zero_vt_pad(_Float16* __restrict__ vt) {
    int i = blockIdx.x * blockDim.x + threadIdx.x;   // over B_*C_*16
    int bo = i >> 4;
    int j  = i & 15;
    vt[(size_t)bo * LP2 + 80 + j] = (_Float16)0.f;
}

// One block per (b, group): reduce 16 channels * 4096 contiguous floats.
__global__ void cab_gn_stats(const float* __restrict__ x, float* __restrict__ stats) {
    int bg = blockIdx.x;
    const float* p = x + (size_t)bg * (GC * HW_);
    float s = 0.f, s2 = 0.f;
    for (int i = threadIdx.x; i < GC * HW_; i += 256) { float v = p[i]; s += v; s2 += v * v; }
    __shared__ float rs[256], rq[256];
    rs[threadIdx.x] = s; rq[threadIdx.x] = s2;
    __syncthreads();
    for (int o = 128; o > 0; o >>= 1) {
        if (threadIdx.x < o) { rs[threadIdx.x] += rs[threadIdx.x + o]; rq[threadIdx.x] += rq[threadIdx.x + o]; }
        __syncthreads();
    }
    if (threadIdx.x == 0) {
        const float inv_n = 1.f / (GC * HW_);
        float mu  = rs[0] * inv_n;
        float var = rq[0] * inv_n - mu * mu;
        stats[bg * 2 + 0] = mu;
        stats[bg * 2 + 1] = rsqrtf(var + 1e-6f);
    }
}

// Normalize + transpose [B,C,HW] f32 -> [B,HW,C] f16 via 32x32 LDS tiles.
__global__ void cab_gn_apply_t(const float* __restrict__ x, const float* __restrict__ stats,
                               const float* __restrict__ gamma, const float* __restrict__ beta,
                               _Float16* __restrict__ xn) {
    __shared__ float tile[32][33];
    int b  = blockIdx.z;
    int c0 = blockIdx.y * 32, p0 = blockIdx.x * 32;
    int tx = threadIdx.x & 31, ty = threadIdx.x >> 5;   // ty 0..7
#pragma unroll
    for (int i = 0; i < 4; ++i) {
        int c = c0 + ty + i * 8;
        int g = c >> 4;
        float mu = stats[(b * 32 + g) * 2], rstd = stats[(b * 32 + g) * 2 + 1];
        float v  = x[((size_t)b * C_ + c) * HW_ + p0 + tx];
        tile[ty + i * 8][tx] = (v - mu) * rstd * gamma[c] + beta[c];
    }
    __syncthreads();
#pragma unroll
    for (int i = 0; i < 4; ++i) {
        int p = p0 + ty + i * 8;
        int c = c0 + tx;
        xn[((size_t)b * HW_ + p) * C_ + c] = (_Float16)tile[tx][ty + i * 8];
    }
}

// Q projection: A = xn [B,HW,C], Bt = q_w [o][c]; wave does 32(M) x 64(N).
// Ping-pong LDS weight panels (async staging overlapped with compute); A and B double-buffered.
__global__ void cab_gemm_qproj(const _Float16* __restrict__ xn, const _Float16* __restrict__ w,
                               const float* __restrict__ bias, _Float16* __restrict__ out) {
    __shared__ __align__(16) _Float16 wpan[2][64 * 256];
    int b  = blockIdx.z;
    int wv = threadIdx.x >> 5;
    int p0 = blockIdx.x * 256 + wv * 32;
    int n0 = blockIdx.y * 64;
    int lane = threadIdx.x & 31, half = lane >> 4, n15 = lane & 15;
    const _Float16* A = xn + ((size_t)b * HW_ + p0) * C_;
    v8f acc[2][4];
#pragma unroll
    for (int g2 = 0; g2 < 2; ++g2)
#pragma unroll
        for (int t = 0; t < 4; ++t) acc[g2][t] = (v8f){0.f,0.f,0.f,0.f,0.f,0.f,0.f,0.f};

    stage_async(wpan[0], w + (size_t)n0 * C_, C_, 256, 2048, 32);
    wait_async_barrier();
    stage_async(wpan[1], w + (size_t)n0 * C_ + 256, C_, 256, 2048, 32);  // overlaps half-0 compute

    for (int kh = 0; kh < 2; ++kh) {
        const _Float16* P = wpan[kh];
        v16h a0 = frag16(A + kh * 256, C_);
        v16h a1 = frag16(A + (size_t)16 * C_ + kh * 256, C_);
#pragma unroll
        for (int kc = 0; kc < 256; kc += 32) {
            v16h na0 = a0, na1 = a1;
            if (kc + 32 < 256) {
                na0 = frag16(A + kh * 256 + kc + 32, C_);
                na1 = frag16(A + (size_t)16 * C_ + kh * 256 + kc + 32, C_);
            }
            v16h bf = frag16(P + kc, 256);
#pragma unroll
            for (int nt = 0; nt < 4; ++nt) {
                v16h nbf = bf;
                if (nt < 3) nbf = frag16(P + (nt + 1) * 16 * 256 + kc, 256);   // B prefetch
                acc[0][nt] = wmma16(a0, bf, acc[0][nt]);
                acc[1][nt] = wmma16(a1, bf, acc[1][nt]);
                bf = nbf;
            }
            a0 = na0; a1 = na1;
        }
        if (kh == 0) wait_async_barrier();   // panel 1 ready
    }
#pragma unroll
    for (int g2 = 0; g2 < 2; ++g2)
#pragma unroll
        for (int nt = 0; nt < 4; ++nt) {
            int o = n0 + nt * 16 + n15;
            float bb = bias[o];
#pragma unroll
            for (int r = 0; r < 8; ++r) {
                int p = p0 + g2 * 16 + r + 8 * half;
                out[((size_t)b * HW_ + p) * C_ + o] = (_Float16)(acc[g2][nt][r] + bb);
            }
        }
}

// K/V projection: A = cond [B,77,768] f32 (row-clamped).
// waves 0..3 -> K matrix, waves 4..7 -> V matrix; each wave one 16x16 tile.
__global__ void cab_gemm_kv(const float* __restrict__ cond,
                            const _Float16* __restrict__ kw, const float* __restrict__ kb,
                            const _Float16* __restrict__ vw, const float* __restrict__ vb,
                            _Float16* __restrict__ kbuf, _Float16* __restrict__ vt) {
    int b  = blockIdx.z;
    int wv = threadIdx.x >> 5;
    int isv = wv >> 2;
    int p0 = blockIdx.x * 16;                       // l-tile (grid.x = 5)
    int n0 = blockIdx.y * 64 + (wv & 3) * 16;       // grid.y = 8
    int lane = threadIdx.x & 31, half = lane >> 4, n15 = lane & 15;
    const float* A = cond + (size_t)b * L_ * CD_ + (size_t)p0 * CD_;
    int rows_left = L_ - p0;
    const _Float16* w   = isv ? vw : kw;
    const float*    bia = isv ? vb : kb;
    v8f acc = (v8f){0.f,0.f,0.f,0.f,0.f,0.f,0.f,0.f};
    for (int kc = 0; kc < CD_; kc += 32) {
        v16h a  = frag16_f32c(A + kc, CD_, rows_left);
        v16h bf = frag16(w + (size_t)n0 * CD_ + kc, CD_);
        acc = wmma16(a, bf, acc);
    }
    int o = n0 + n15;
    float bb = bia[o];
#pragma unroll
    for (int r = 0; r < 8; ++r) {
        int l = p0 + r + 8 * half;                  // < 80 ; rows 77..79 masked later
        float v = acc[r] + bb;
        if (!isv) kbuf[((size_t)b * LP + l) * C_ + o]  = (_Float16)v;
        else      vt  [((size_t)b * C_ + o) * LP2 + l] = (_Float16)v;
    }
}

// Attention: block = one (b,head), 128 query rows; K/Vt panels staged in LDS (async).
__global__ void cab_attn(const _Float16* __restrict__ q, const _Float16* __restrict__ kbuf,
                         const _Float16* __restrict__ vt, const float* __restrict__ scale_p,
                         _Float16* __restrict__ obuf) {
    __shared__ __align__(16) _Float16 kls[LP * HD];        // 80 x 64 = 10KB
    __shared__ __align__(16) _Float16 vls[HD * LP2];       // 64 x 96 = 12KB
    __shared__ __align__(16) _Float16 lds_p[8][16 * LP2];  // 24KB
    int b = blockIdx.z, h = blockIdx.y;
    int wv = threadIdx.x >> 5;
    int p0 = blockIdx.x * 128 + wv * 16;
    int lane = threadIdx.x & 31, half = lane >> 4, n15 = lane & 15;
    float scl = scale_p[0];

    // Stage K (rows l, ld C_) and Vt (rows d, ld LP2) panels for this (b,h).
    stage_async(kls, kbuf + (size_t)b * LP * C_ + h * HD, C_, HD, (LP * HD) / 8, HD / 8);
    stage_async(vls, vt + ((size_t)b * C_ + h * HD) * LP2, LP2, LP2, (HD * LP2) / 8, LP2 / 8);

    const _Float16* Q = q + ((size_t)b * HW_ + p0) * C_ + h * HD;
    v16h a0 = frag16(Q,      C_);
    v16h a1 = frag16(Q + 32, C_);

    wait_async_barrier();

    v8f s[5];
#pragma unroll
    for (int t = 0; t < 5; ++t) {
        v8f z = (v8f){0.f,0.f,0.f,0.f,0.f,0.f,0.f,0.f};
        z = wmma16(a0, frag16(kls + (t * 16) * HD,      HD), z);
        z = wmma16(a1, frag16(kls + (t * 16) * HD + 32, HD), z);
        s[t] = z;
    }

    _Float16* P = &lds_p[wv][0];
    bool mask4 = (n15 >= 13);                       // t==4 column l = 64+n15 >= 77
#pragma unroll
    for (int r = 0; r < 8; ++r) {
        float sv[5], m = -3.0e38f;
#pragma unroll
        for (int t = 0; t < 5; ++t) {
            float v = s[t][r] * scl;
            if (t == 4 && mask4) v = -3.0e38f;
            sv[t] = v; m = fmaxf(m, v);
        }
#pragma unroll
        for (int o = 1; o < 16; o <<= 1) m = fmaxf(m, __shfl_xor(m, o, 32));
        float sum = 0.f, pv[5];
#pragma unroll
        for (int t = 0; t < 5; ++t) { float e = __expf(sv[t] - m); pv[t] = e; sum += e; }
#pragma unroll
        for (int o = 1; o < 16; o <<= 1) sum += __shfl_xor(sum, o, 32);
        float inv = 1.f / sum;
        int row = r + 8 * half;
#pragma unroll
        for (int t = 0; t < 5; ++t) P[row * LP2 + t * 16 + n15] = (_Float16)(pv[t] * inv);
        P[row * LP2 + 80 + n15] = (_Float16)0.f;    // zero pad cols 80..95
    }
    __syncthreads();

    v8f oacc[4];
#pragma unroll
    for (int nt = 0; nt < 4; ++nt) oacc[nt] = (v8f){0.f,0.f,0.f,0.f,0.f,0.f,0.f,0.f};
#pragma unroll
    for (int kc = 0; kc < 3; ++kc) {
        v16h ap = frag16(P + kc * 32, LP2);         // A-frag of probabilities from LDS
#pragma unroll
        for (int nt = 0; nt < 4; ++nt)
            oacc[nt] = wmma16(ap, frag16(vls + (nt * 16) * LP2 + kc * 32, LP2), oacc[nt]);
    }
#pragma unroll
    for (int nt = 0; nt < 4; ++nt)
#pragma unroll
        for (int r = 0; r < 8; ++r) {
            int p = p0 + r + 8 * half;
            obuf[((size_t)b * HW_ + p) * C_ + h * HD + nt * 16 + n15] = (_Float16)oacc[nt][r];
        }
}

// Final projection + residual: out[b,o,p] = x + O @ proj_w^T + proj_b.
// Ping-pong LDS weight panels; low region reused as f32 transpose buffer for coalesced stores.
__global__ void cab_gemm_proj(const _Float16* __restrict__ obuf, const _Float16* __restrict__ w,
                              const float* __restrict__ bias, const float* __restrict__ x,
                              float* __restrict__ out) {
    __shared__ __align__(16) char smem[65536];             // 2 x 32KB panels; reused for transpose
    _Float16* wpan0 = (_Float16*)smem;
    _Float16* wpan1 = (_Float16*)(smem + 32768);
    int b  = blockIdx.z;
    int wv = threadIdx.x >> 5;
    int p0 = blockIdx.x * 128 + wv * 16;
    int n0 = blockIdx.y * 64;
    int lane = threadIdx.x & 31, half = lane >> 4, n15 = lane & 15;
    const _Float16* A = obuf + ((size_t)b * HW_ + p0) * C_;
    v8f acc[4];
#pragma unroll
    for (int t = 0; t < 4; ++t) acc[t] = (v8f){0.f,0.f,0.f,0.f,0.f,0.f,0.f,0.f};

    stage_async(wpan0, w + (size_t)n0 * C_, C_, 256, 2048, 32);
    wait_async_barrier();
    stage_async(wpan1, w + (size_t)n0 * C_ + 256, C_, 256, 2048, 32);    // overlaps half-0 compute

    for (int kh = 0; kh < 2; ++kh) {
        const _Float16* P = kh ? wpan1 : wpan0;
        v16h a = frag16(A + kh * 256, C_);
#pragma unroll
        for (int kc = 0; kc < 256; kc += 32) {
            v16h na = a;
            if (kc + 32 < 256) na = frag16(A + kh * 256 + kc + 32, C_);
            v16h bf = frag16(P + kc, 256);
#pragma unroll
            for (int nt = 0; nt < 4; ++nt) {
                v16h nbf = bf;
                if (nt < 3) nbf = frag16(P + (nt + 1) * 16 * 256 + kc, 256);  // B prefetch
                acc[nt] = wmma16(a, bf, acc[nt]);
                bf = nbf;
            }
            a = na;
        }
        if (kh == 0) wait_async_barrier();   // panel 1 ready
    }
    __syncthreads();                          // all weight reads done before smem reuse

    float* T = (float*)smem + (size_t)wv * 64 * 17;        // [64 o][16 p], stride 17
#pragma unroll
    for (int nt = 0; nt < 4; ++nt)
#pragma unroll
        for (int r = 0; r < 8; ++r)
            T[(nt * 16 + n15) * 17 + r + 8 * half] = acc[nt][r];
    __syncthreads();
#pragma unroll
    for (int i = 0; i < 32; ++i) {
        int orow = i * 2 + half;
        int o = n0 + orow;
        size_t idx = ((size_t)b * C_ + o) * HW_ + p0 + n15;
        out[idx] = x[idx] + T[orow * 17 + n15] + bias[o];
    }
}

extern "C" void kernel_launch(void* const* d_in, const int* in_sizes, int n_in,
                              void* d_out, int out_size, void* d_ws, size_t ws_size,
                              hipStream_t stream) {
    const float* x      = (const float*)d_in[0];
    const float* cond   = (const float*)d_in[1];
    const float* gamma  = (const float*)d_in[2];
    const float* beta   = (const float*)d_in[3];
    const float* q_w    = (const float*)d_in[4];
    const float* q_b    = (const float*)d_in[5];
    const float* k_w    = (const float*)d_in[6];
    const float* k_b    = (const float*)d_in[7];
    const float* v_w    = (const float*)d_in[8];
    const float* v_b    = (const float*)d_in[9];
    const float* proj_w = (const float*)d_in[10];
    const float* proj_b = (const float*)d_in[11];
    const float* scale  = (const float*)d_in[12];
    float* out = (float*)d_out;

    char* ws = (char*)d_ws;
    size_t off = 0;
    auto carve = [&](size_t bytes) { void* p = ws + off; off = (off + bytes + 255) & ~(size_t)255; return p; };
    float*     stats = (float*)    carve(512 * 2 * sizeof(float));
    _Float16*  qw16  = (_Float16*) carve((size_t)C_ * C_  * 2);
    _Float16*  kw16  = (_Float16*) carve((size_t)C_ * CD_ * 2);
    _Float16*  vw16  = (_Float16*) carve((size_t)C_ * CD_ * 2);
    _Float16*  pw16  = (_Float16*) carve((size_t)C_ * C_  * 2);
    _Float16*  kbuf  = (_Float16*) carve((size_t)B_ * LP  * C_  * 2);
    _Float16*  vt    = (_Float16*) carve((size_t)B_ * C_  * LP2 * 2);
    _Float16*  xn    = (_Float16*) carve((size_t)B_ * HW_ * C_  * 2);  // reused as obuf after qproj
    _Float16*  qbuf  = (_Float16*) carve((size_t)B_ * HW_ * C_  * 2);
    _Float16*  obuf  = xn;   // xn is dead once qproj finishes (stream-ordered)

    // 1) weights to f16
    cab_cvt_f16<<<(C_ * C_  + 255) / 256, 256, 0, stream>>>(q_w,    qw16, C_ * C_);
    cab_cvt_f16<<<(C_ * CD_ + 255) / 256, 256, 0, stream>>>(k_w,    kw16, C_ * CD_);
    cab_cvt_f16<<<(C_ * CD_ + 255) / 256, 256, 0, stream>>>(v_w,    vw16, C_ * CD_);
    cab_cvt_f16<<<(C_ * C_  + 255) / 256, 256, 0, stream>>>(proj_w, pw16, C_ * C_);

    // 2) GroupNorm
    cab_gn_stats<<<B_ * 32, 256, 0, stream>>>(x, stats);
    cab_gn_apply_t<<<dim3(HW_ / 32, C_ / 32, B_), 256, 0, stream>>>(x, stats, gamma, beta, xn);

    // 3) projections
    cab_gemm_qproj<<<dim3(HW_ / 256, C_ / 64, B_), 256, 0, stream>>>(xn, qw16, q_b, qbuf);
    cab_gemm_kv<<<dim3(LP / 16, C_ / 64, B_), 256, 0, stream>>>(cond, kw16, k_b, vw16, v_b, kbuf, vt);
    cab_zero_vt_pad<<<(B_ * C_ * 16) / 256, 256, 0, stream>>>(vt);

    // 4) attention (writes obuf == xn region)
    cab_attn<<<dim3(HW_ / 128, NH, B_), 256, 0, stream>>>(qbuf, kbuf, vt, scale, obuf);

    // 5) output projection + residual
    cab_gemm_proj<<<dim3(HW_ / 128, C_ / 64, B_), 256, 0, stream>>>(obuf, pw16, proj_b, x, out);
}
